// MACEModel_29815662969336
// MI455X (gfx1250) — matrix operand order repaired and checked by reading, not compiled, not run on hardware
//
#include <hip/hip_runtime.h>
#include <hip/hip_bf16.h>

#define N_NODES 512
#define N_EDGES 4096
#define N_GRAPHS 32
#define EMB 64
#define HSTR 576   // 64 channels * 9 spherical components per node

typedef __attribute__((ext_vector_type(16))) __bf16 v16bf;
typedef __attribute__((ext_vector_type(8)))  __bf16 v8bf;
typedef __attribute__((ext_vector_type(8)))  float  v8f;

union V16 { v16bf v; v8bf h[2]; };

// ---------- helpers ----------
__device__ __forceinline__ __bf16 f2bf(float f) { return (__bf16)f; }

__device__ __forceinline__ unsigned int pack_bf2(float a, float b) {
  __bf16 x = (__bf16)a, y = (__bf16)b;
  unsigned short ux, uy;
  __builtin_memcpy(&ux, &x, 2);
  __builtin_memcpy(&uy, &y, 2);
  return (unsigned int)ux | ((unsigned int)uy << 16);
}

struct PathInfo { short l1, l2, l3, cg; float norm; };

// CONV_PATHS[0]: (0,0,0)(0,1,1)(0,2,2); norm = 1/sqrt(1*64)
__constant__ PathInfo g_paths0[3] = {
  {0,0,0,0, 0.125f}, {0,1,1,1, 0.125f}, {0,2,2,2, 0.125f}
};
// CONV_PATHS[1] == PROD_PATHS triples; norms = 1/sqrt(n_in[l3]*64), n_in={0:3,1:4,2:4}
__constant__ PathInfo g_paths1[11] = {
  {0,0,0,0, 0.07216878365f},
  {0,1,1,1, 0.0625f},
  {0,2,2,2, 0.0625f},
  {1,0,1,3, 0.0625f},
  {1,1,0,4, 0.07216878365f},
  {1,1,2,5, 0.0625f},
  {1,2,1,6, 0.0625f},
  {2,0,2,7, 0.0625f},
  {2,1,1,8, 0.0625f},
  {2,2,0,9, 0.07216878365f},
  {2,2,2,10,0.0625f}
};
__constant__ short g_trip[11][3] = {
  {0,0,0},{0,1,1},{0,2,2},{1,0,1},{1,1,0},{1,1,2},{1,2,1},{2,0,2},{2,1,1},{2,2,0},{2,2,2}
};

// ---------- CG tensor init (on device; deterministic; capture-safe) ----------
__device__ double dfact(int n) { double r = 1.0; for (int i = 2; i <= n; i++) r *= (double)i; return r; }

__device__ double cg_coeff(int j1,int m1,int j2,int m2,int j3,int m3) {
  if (m1 + m2 != m3) return 0.0;
  int lo = (j1 > j2) ? (j1 - j2) : (j2 - j1);
  if (j3 < lo || j3 > j1 + j2) return 0.0;
  double pref = sqrt((double)(2*j3+1) * dfact(j1+j2-j3) * dfact(j1-j2+j3) * dfact(-j1+j2+j3) / dfact(j1+j2+j3+1));
  pref *= sqrt(dfact(j1+m1)*dfact(j1-m1)*dfact(j2+m2)*dfact(j2-m2)*dfact(j3+m3)*dfact(j3-m3));
  double s = 0.0;
  for (int k = 0; k <= j1 + j2 + j3; k++) {
    int d0=k, d1=j1+j2-j3-k, d2=j1-m1-k, d3=j2+m2-k, d4=j3-j2+m1+k, d5=j3-j1-m2+k;
    if (d0<0||d1<0||d2<0||d3<0||d4<0||d5<0) continue;
    double pr = dfact(d0)*dfact(d1)*dfact(d2)*dfact(d3)*dfact(d4)*dfact(d5);
    s += ((k & 1) ? -1.0 : 1.0) / pr;
  }
  return pref * s;
}

__device__ void real_U(int l, double Ure[5][5], double Uim[5][5]) {
  for (int a = 0; a < 5; a++) for (int b = 0; b < 5; b++) { Ure[a][b] = 0.0; Uim[a][b] = 0.0; }
  Ure[l][l] = 1.0;
  const double is2 = 0.7071067811865476;
  for (int m = 1; m <= l; m++) {
    double sg = (m & 1) ? -1.0 : 1.0;
    Ure[l+m][l+m] = sg * is2;
    Ure[l+m][l-m] = is2;
    Uim[l-m][l-m] = is2;
    Uim[l-m][l+m] = -sg * is2;
  }
}

// one block per triple; 128 threads; threads 0..124 compute one (a,b,c) element
__global__ void cg_init_kernel(float* __restrict__ cg) {
  int t = blockIdx.x, tid = threadIdx.x;
  int l1 = g_trip[t][0], l2 = g_trip[t][1], l3 = g_trip[t][2];
  int d1 = 2*l1+1, d2 = 2*l2+1, d3 = 2*l3+1;
  __shared__ float sre[125], sim_[125];
  __shared__ int useImag;
  int a = tid / 25, b = (tid / 5) % 5, c = tid % 5;
  bool active = (tid < 125) && (a < d1) && (b < d2) && (c < d3);
  double re = 0.0, im = 0.0;
  if (active) {
    double U1r[5][5], U1i[5][5], U2r[5][5], U2i[5][5], U3r[5][5], U3i[5][5];
    real_U(l1, U1r, U1i); real_U(l2, U2r, U2i); real_U(l3, U3r, U3i);
    for (int i = 0; i < d1; i++)
      for (int j = 0; j < d2; j++)
        for (int k = 0; k < d3; k++) {
          double cc = cg_coeff(l1, i - l1, l2, j - l2, l3, k - l3);
          if (cc == 0.0) continue;
          double pr = U1r[a][i]*U2r[b][j] - U1i[a][i]*U2i[b][j];
          double pi = U1r[a][i]*U2i[b][j] + U1i[a][i]*U2r[b][j];
          double qr = pr*U3r[c][k] + pi*U3i[c][k];   // * conj(U3)
          double qi = pi*U3r[c][k] - pr*U3i[c][k];
          re += qr * cc; im += qi * cc;
        }
  }
  if (tid < 125) { sre[tid] = (float)fabs(re); sim_[tid] = (float)fabs(im); }
  __syncthreads();
  if (tid == 0) {
    float mr = 0.f, mi = 0.f;
    for (int i = 0; i < 125; i++) { mr = fmaxf(mr, sre[i]); mi = fmaxf(mi, sim_[i]); }
    useImag = (mi > mr) ? 1 : 0;
  }
  __syncthreads();
  if (tid < 125) cg[t * 125 + tid] = (float)(useImag ? im : re);
}

// ---------- geometry: vec, spherical harmonics, Bessel radial basis ----------
__global__ void edge_geom_kernel(const float* __restrict__ pos, const int* __restrict__ src,
                                 const int* __restrict__ dst, float* __restrict__ sh,
                                 float* __restrict__ ef) {
  int e = blockIdx.x * blockDim.x + threadIdx.x;
  if (e >= N_EDGES) return;
  int s = src[e], d = dst[e];
  float vx = pos[s*3+0] - pos[d*3+0];
  float vy = pos[s*3+1] - pos[d*3+1];
  float vz = pos[s*3+2] - pos[d*3+2];
  float r = sqrtf(vx*vx + vy*vy + vz*vz);
  float inv = 1.0f / r;
  float x = vx*inv, y = vy*inv, z = vz*inv;
  const float s3 = 1.7320508075688772f, s5 = 2.2360679774997896f, s15 = 3.872983346207417f;
  float* o = sh + e*9;
  o[0] = 1.0f;
  o[1] = s3*y; o[2] = s3*z; o[3] = s3*x;
  o[4] = s15*x*y; o[5] = s15*y*z; o[6] = 0.5f*s5*(3.f*z*z - 1.f);
  o[7] = s15*x*z; o[8] = 0.5f*s15*(x*x - y*y);
  float u = r * 0.1f;  // r / R_MAX
  float env = 0.f;
  if (u < 1.f) {
    float u5 = u*u*u*u*u;
    env = 1.f - 21.f*u5 + 35.f*u5*u - 15.f*u5*u*u;   // P_CUT=5 polynomial cutoff
  }
  float pref = 0.44721359549995794f * env * inv;      // sqrt(2/R_MAX)
  #pragma unroll
  for (int n = 1; n <= 8; n++)
    ef[e*8 + n - 1] = pref * sinf((float)n * 3.14159265358979323846f * u);
}

__global__ void h_init_kernel(const int* __restrict__ atoms, const float* __restrict__ embed,
                              float* __restrict__ h) {
  int idx = blockIdx.x * blockDim.x + threadIdx.x;
  if (idx >= N_NODES * EMB) return;
  int n = idx / EMB, u = idx % EMB;
  float* o = h + n*HSTR + u*9;
  o[0] = embed[atoms[n]*EMB + u];
  #pragma unroll
  for (int k = 1; k < 9; k++) o[k] = 0.f;
}

__global__ void edge_mlp_kernel(const float* __restrict__ ef, const float* __restrict__ fc1w,
                                const float* __restrict__ fc1b, float* __restrict__ A) {
  int idx = blockIdx.x * blockDim.x + threadIdx.x;
  if (idx >= N_EDGES * EMB) return;
  int e = idx / EMB, c = idx % EMB;
  float s = fc1b[c];
  #pragma unroll
  for (int n = 0; n < 8; n++) s += ef[e*8 + n] * fc1w[n*EMB + c];
  A[idx] = fmaxf(s, 0.f);
}

__global__ void zero_kernel(float* __restrict__ p, int n) {
  int i = blockIdx.x * blockDim.x + threadIdx.x;
  if (i < n) p[i] = 0.f;
}

// ---------- pre-swizzle fc2_w into bf16 B-fragment order ----------
// out[o]: o = ((((i*64+u)*4+vt)*2+kk)*32+lane)*16+t
// value  = fc2_w[c][i*4096 + u*64 + vt*16 + v], c = (lane<16?0:8)+kk*32+(t<8?t:t+8), v = lane&15.
// Hot loop then reads one 32B contiguous v16bf per fragment per lane.
__global__ void fc2_swizzle_kernel(const float* __restrict__ fc2w, __bf16* __restrict__ bsw,
                                   int npaths) {
  int o = blockIdx.x * blockDim.x + threadIdx.x;
  if (o >= npaths * 262144) return;
  int t    = o & 15;
  int lane = (o >> 4) & 31;
  int kk   = (o >> 9) & 1;
  int vt   = (o >> 10) & 3;
  int u    = (o >> 12) & 63;
  int i    = o >> 18;
  int c = ((lane < 16) ? 0 : 8) + kk*32 + ((t < 8) ? t : t + 8);
  int v = lane & 15;
  long cstride = (long)npaths * 4096;
  bsw[o] = (__bf16)fc2w[(long)c * cstride + (long)i*4096 + u*64 + vt*16 + v];
}

// ---------- the heavy WMMA kernel ----------
// block = 128 threads (4 wave32), handles 16 edges x 1 path.
// stage 1: W[16e][16v][64u] = A[16e x 64c] @ fc2_w[64c x (u,v)] + bias  (bf16 WMMA, f32 acc)
// stage 2: msg[e][16v][k]   = sum_u W[e][v][u] * t[e][k][u]             (bf16 WMMA per edge)
// scatter: atomicAdd into hu[dst][v][l3-block] with path normalization.
__global__ __launch_bounds__(128) void conv_kernel(
    const float* __restrict__ h_in, const float* __restrict__ sh,
    const float* __restrict__ A,    const __bf16* __restrict__ bsw,
    const float* __restrict__ fc2b, const float* __restrict__ cg,
    const int* __restrict__ src,    const int* __restrict__ dst,
    float* __restrict__ hu, int layer, int npaths)
{
  extern __shared__ char smem[];
  __bf16* sA = (__bf16*)smem;                       // [16e][64c]       2 KB
  __bf16* sT = (__bf16*)(smem + 2048);              // [16e][8k][64u]  16 KB
  __bf16* sW = (__bf16*)(smem + 2048 + 16384);      // [16e][16v][64u] 32 KB

  const int tid  = threadIdx.x;
  const int lane = tid & 31;
  const int wave = tid >> 5;
  const int e0   = blockIdx.x * 16;
  const int ip   = blockIdx.y;
  PathInfo P = (layer == 0) ? g_paths0[ip] : g_paths1[ip];
  const int l1 = P.l1, l2 = P.l2, l3 = P.l3;
  const int dk1 = 2*l1+1, dk2 = 2*l2+1, dk3 = 2*l3+1;
  const int koff3 = l3 * l3;
  const float* CGp = cg + (int)P.cg * 125;
  const long pbase = (long)ip * 4096;
  const __bf16* bswp = bsw + (long)ip * 262144;

  // fill sA (bf16 radial activations) and sT[e][k][u] (CG-contracted tensor, k rows 0..7)
  for (int idx = tid; idx < 16 * EMB; idx += 128) {
    int e = idx >> 6, u = idx & 63;
    int ge = e0 + e;
    sA[idx] = f2bf(A[ge * EMB + u]);
    int sn = src[ge];
    float hv[5], shv[5];
    #pragma unroll
    for (int i = 0; i < 5; i++) hv[i]  = (i < dk1) ? h_in[sn*HSTR + u*9 + l1*l1 + i] : 0.f;
    #pragma unroll
    for (int j = 0; j < 5; j++) shv[j] = (j < dk2) ? sh[ge*9 + l2*l2 + j] : 0.f;
    #pragma unroll
    for (int k = 0; k < 8; k++) {
      float t = 0.f;
      if (k < dk3)
        for (int i = 0; i < dk1; i++)
          for (int j = 0; j < dk2; j++)
            t += hv[i] * shv[j] * CGp[(i*5 + j)*5 + k];
      sT[(e*8 + k)*64 + u] = f2bf(t);
    }
  }
  __syncthreads();

  // hoisted stage-1 A-fragments (M=edge, K=c): two contiguous 8-element runs per kk
  const int hb  = (lane < 16) ? 0 : 8;
  const int h16 = (lane < 16) ? 0 : 16;
  v16bf aA[2];
  {
    int e = lane & 15;
    #pragma unroll
    for (int kk = 0; kk < 2; kk++) {
      V16 tmp;
      tmp.h[0] = *(const v8bf*)(sA + e*64 + hb + kk*32);
      tmp.h[1] = *(const v8bf*)(sA + e*64 + hb + kk*32 + 16);
      aA[kk] = tmp.v;
    }
  }
  v16bf zfrag;
  #pragma unroll
  for (int t = 0; t < 16; t++) zfrag[t] = (__bf16)0.0f;

  const int vlane = lane & 15;   // N index: v for stage1/2-A, k for stage2-B
  for (int vt = 0; vt < 4; vt++) {
    const int v0 = vt * 16;
    // ---- stage 1: two independent u-columns in flight per wave ----
    for (int uu = 0; uu < 16; uu += 2) {
      const int u0 = wave*16 + uu;
      v8f acc0 = {}, acc1 = {};
      #pragma unroll
      for (int kk = 0; kk < 2; kk++) {
        const __bf16* p0 = bswp + ((((long)u0*4 + vt)*2 + kk)*32 + lane)*16;
        v16bf b0 = *(const v16bf*)p0;
        v16bf b1 = *(const v16bf*)(p0 + 4096);  // u0+1 (stride per u = 4*2*32*16)
        acc0 = __builtin_amdgcn_wmma_f32_16x16x32_bf16(false, aA[kk], false, b0,
                                                       (short)0, acc0, false, false);
        acc1 = __builtin_amdgcn_wmma_f32_16x16x32_bf16(false, aA[kk], false, b1,
                                                       (short)0, acc1, false, false);
      }
      float bias0 = fc2b[pbase + u0*64 + v0 + vlane];
      float bias1 = fc2b[pbase + (u0+1)*64 + v0 + vlane];
      int ebase = (lane < 16) ? 0 : 8;
      #pragma unroll
      for (int r = 0; r < 8; r++) {
        int e = r + ebase;
        *(unsigned int*)(sW + (e*16 + vlane)*64 + u0) =
            pack_bf2(acc0[r] + bias0, acc1[r] + bias1);
      }
    }
    __syncthreads();
    // ---- stage 2: two independent edges in flight per wave ----
    #pragma unroll
    for (int half = 0; half < 2; half++) {
      const int ea = wave + half*8;
      const int eb = ea + 4;
      v8f accA = {}, accB = {};
      #pragma unroll
      for (int kk = 0; kk < 2; kk++) {
        V16 wa, wb;
        wa.h[0] = *(const v8bf*)(sW + (ea*16 + vlane)*64 + h16 + kk*32);
        wa.h[1] = *(const v8bf*)(sW + (ea*16 + vlane)*64 + h16 + kk*32 + 8);
        wb.h[0] = *(const v8bf*)(sW + (eb*16 + vlane)*64 + h16 + kk*32);
        wb.h[1] = *(const v8bf*)(sW + (eb*16 + vlane)*64 + h16 + kk*32 + 8);
        v16bf ta = zfrag, tb = zfrag;
        if (vlane < 8) {   // only k-rows 0..7 exist; others are zero columns
          ta = *(const v16bf*)(sT + (ea*8 + vlane)*64 + h16 + kk*32);
          tb = *(const v16bf*)(sT + (eb*8 + vlane)*64 + h16 + kk*32);
        }
        accA = __builtin_amdgcn_wmma_f32_16x16x32_bf16(false, wa.v, false, ta,
                                                       (short)0, accA, false, false);
        accB = __builtin_amdgcn_wmma_f32_16x16x32_bf16(false, wb.v, false, tb,
                                                       (short)0, accB, false, false);
      }
      if (vlane < dk3) {
        int gdA = dst[e0 + ea], gdB = dst[e0 + eb];
        int vb = v0 + ((lane < 16) ? 0 : 8);
        #pragma unroll
        for (int r = 0; r < 8; r++) {
          atomicAdd(&hu[gdA*HSTR + (vb + r)*9 + koff3 + vlane], accA[r] * P.norm);
          atomicAdd(&hu[gdB*HSTR + (vb + r)*9 + koff3 + vlane], accB[r] * P.norm);
        }
      }
    }
    __syncthreads();
  }
}

// ---------- product block ----------
__global__ void product1_kernel(const float* __restrict__ x, const float* __restrict__ cg,
                                const float* __restrict__ w1, const float* __restrict__ w2,
                                const float* __restrict__ w3, float* __restrict__ tmp) {
  int idx = blockIdx.x * blockDim.x + threadIdx.x;
  if (idx >= N_NODES * EMB) return;
  int n = idx / EMB, u = idx % EMB;
  float xv[9];
  #pragma unroll
  for (int k = 0; k < 9; k++) xv[k] = x[n*HSTR + u*9 + k];
  float a2[9], a3[9];
  #pragma unroll
  for (int k = 0; k < 9; k++) { a2[k] = 0.f; a3[k] = 0.f; }
  for (int p = 0; p < 11; p++) {           // a2 = couple(x, x, w2)
    PathInfo P = g_paths1[p];
    float wp = w2[p*EMB + u];
    const float* C = cg + (int)P.cg * 125;
    int o1 = P.l1*P.l1, o2 = P.l2*P.l2, o3 = P.l3*P.l3;
    int dk1 = 2*P.l1+1, dk2 = 2*P.l2+1, dk3 = 2*P.l3+1;
    for (int k = 0; k < dk3; k++) {
      float s = 0.f;
      for (int i = 0; i < dk1; i++)
        for (int j = 0; j < dk2; j++)
          s += xv[o1+i] * xv[o2+j] * C[(i*5+j)*5 + k];
      a2[o3 + k] += s * wp;
    }
  }
  const float inv_s3 = 0.5773502691896258f;  // cnt={0:3,1:4,2:4}
  a2[0] *= inv_s3;
  #pragma unroll
  for (int k = 1; k < 9; k++) a2[k] *= 0.5f;
  for (int p = 0; p < 11; p++) {           // a3 = couple(a2, x, w3)
    PathInfo P = g_paths1[p];
    float wp = w3[p*EMB + u];
    const float* C = cg + (int)P.cg * 125;
    int o1 = P.l1*P.l1, o2 = P.l2*P.l2, o3 = P.l3*P.l3;
    int dk1 = 2*P.l1+1, dk2 = 2*P.l2+1, dk3 = 2*P.l3+1;
    for (int k = 0; k < dk3; k++) {
      float s = 0.f;
      for (int i = 0; i < dk1; i++)
        for (int j = 0; j < dk2; j++)
          s += a2[o1+i] * xv[o2+j] * C[(i*5+j)*5 + k];
      a3[o3 + k] += s * wp;
    }
  }
  a3[0] *= inv_s3;
  #pragma unroll
  for (int k = 1; k < 9; k++) a3[k] *= 0.5f;
  float w10 = w1[0*EMB + u], w11 = w1[1*EMB + u], w12 = w1[2*EMB + u];
  float* o = tmp + n*HSTR + u*9;
  o[0] = xv[0]*w10 + a2[0] + a3[0];
  #pragma unroll
  for (int k = 1; k < 4; k++) o[k] = xv[k]*w11 + a2[k] + a3[k];
  #pragma unroll
  for (int k = 4; k < 9; k++) o[k] = xv[k]*w12 + a2[k] + a3[k];
}

__global__ void product2_kernel(const float* __restrict__ tmp, const float* __restrict__ lin,
                                const float* __restrict__ h_prev, float* __restrict__ h_out) {
  int idx = blockIdx.x * blockDim.x + threadIdx.x;
  if (idx >= N_NODES * HSTR) return;
  int n = idx / HSTR, rem = idx % HSTR, v = rem / 9, kk = rem % 9;
  int l = (kk == 0) ? 0 : ((kk < 4) ? 1 : 2);
  const float* lp = lin + l*4096 + v;   // lin[l][u][v], stride 64 over u
  const float* tp = tmp + n*HSTR + kk;  // tmp[n][u][kk], stride 9 over u
  float s = 0.f;
  #pragma unroll 8
  for (int u = 0; u < EMB; u++) s += tp[u*9] * lp[u*64];
  h_out[idx] = s * 0.125f + h_prev[idx];  // 1/sqrt(EMB) + skip
}

// ---------- readout ----------
__global__ void pool_kernel(const float* __restrict__ h, const int* __restrict__ batch,
                            float* __restrict__ pooled) {
  int idx = blockIdx.x * blockDim.x + threadIdx.x;
  if (idx >= N_NODES * EMB) return;
  int n = idx / EMB, u = idx % EMB;
  atomicAdd(&pooled[batch[n]*EMB + u], h[n*HSTR + u*9]);
}

__global__ void readout_kernel(const float* __restrict__ pooled, const float* __restrict__ w1,
                               const float* __restrict__ b1, const float* __restrict__ w2,
                               const float* __restrict__ b2, float* __restrict__ out) {
  int g = blockIdx.x, c = threadIdx.x;
  __shared__ float shm[EMB];
  float s = b1[c];
  for (int u = 0; u < EMB; u++) s += pooled[g*EMB + u] * w1[u*EMB + c];
  s = fmaxf(s, 0.f);
  shm[c] = s * w2[c];
  __syncthreads();
  if (c == 0) {
    float t = b2[0];
    for (int i = 0; i < EMB; i++) t += shm[i];
    out[g] = t;
  }
}

// ---------- host launch ----------
extern "C" void kernel_launch(void* const* d_in, const int* in_sizes, int n_in,
                              void* d_out, int out_size, void* d_ws, size_t ws_size,
                              hipStream_t stream) {
  (void)in_sizes; (void)n_in; (void)out_size; (void)ws_size;
  // setup_inputs order, params flattened dict-key-sorted (jax pytree order):
  const int*   atoms = (const int*)d_in[0];
  const float* pos   = (const float*)d_in[1];
  const int*   eidx  = (const int*)d_in[2];
  const int*   batch = (const int*)d_in[3];
  const float* embed = (const float*)d_in[4];
  const float* fc1b[2] = {(const float*)d_in[5],  (const float*)d_in[13]};
  const float* fc1w[2] = {(const float*)d_in[6],  (const float*)d_in[14]};
  const float* fc2b[2] = {(const float*)d_in[7],  (const float*)d_in[15]};
  const float* fc2w[2] = {(const float*)d_in[8],  (const float*)d_in[16]};
  const float* linp[2] = {(const float*)d_in[9],  (const float*)d_in[17]};
  const float* w1p [2] = {(const float*)d_in[10], (const float*)d_in[18]};
  const float* w2p [2] = {(const float*)d_in[11], (const float*)d_in[19]};
  const float* w3p [2] = {(const float*)d_in[12], (const float*)d_in[20]};
  const float* p1b = (const float*)d_in[21];
  const float* p1w = (const float*)d_in[22];
  const float* p2b = (const float*)d_in[23];
  const float* p2w = (const float*)d_in[24];
  const int* src = eidx;
  const int* dst = eidx + N_EDGES;

  float* ws   = (float*)d_ws;
  float* CG   = ws;                       // 11*125 -> pad 1408
  float* SH   = CG  + 1408;               // E*9
  float* EF   = SH  + N_EDGES*9;          // E*8
  float* Abuf = EF  + N_EDGES*8;          // E*64
  float* hA   = Abuf+ N_EDGES*EMB;        // N*576
  float* hB   = hA  + N_NODES*HSTR;
  float* HU   = hB  + N_NODES*HSTR;
  float* TMP  = HU  + N_NODES*HSTR;
  float* POOL = TMP + N_NODES*HSTR;       // 32*64
  __bf16* BSW = (__bf16*)(POOL + N_GRAPHS*EMB);  // up to 11*262144 bf16 (32B-aligned)

  cg_init_kernel<<<11, 128, 0, stream>>>(CG);
  edge_geom_kernel<<<(N_EDGES + 255)/256, 256, 0, stream>>>(pos, src, dst, SH, EF);
  h_init_kernel<<<(N_NODES*EMB + 255)/256, 256, 0, stream>>>(atoms, embed, hA);

  float* hprev = hA; float* hnext = hB;
  for (int L = 0; L < 2; L++) {
    int npaths = (L == 0) ? 3 : 11;
    fc2_swizzle_kernel<<<(npaths*262144 + 255)/256, 256, 0, stream>>>(fc2w[L], BSW, npaths);
    edge_mlp_kernel<<<(N_EDGES*EMB + 255)/256, 256, 0, stream>>>(EF, fc1w[L], fc1b[L], Abuf);
    zero_kernel<<<(N_NODES*HSTR + 255)/256, 256, 0, stream>>>(HU, N_NODES*HSTR);
    dim3 grid(N_EDGES/16, npaths);
    conv_kernel<<<grid, 128, 51200, stream>>>(hprev, SH, Abuf, BSW, fc2b[L], CG,
                                              src, dst, HU, L, npaths);
    product1_kernel<<<(N_NODES*EMB + 255)/256, 256, 0, stream>>>(HU, CG, w1p[L], w2p[L], w3p[L], TMP);
    product2_kernel<<<(N_NODES*HSTR + 255)/256, 256, 0, stream>>>(TMP, linp[L], hprev, hnext);
    float* t = hprev; hprev = hnext; hnext = t;
  }
  zero_kernel<<<(N_GRAPHS*EMB + 255)/256, 256, 0, stream>>>(POOL, N_GRAPHS*EMB);
  pool_kernel<<<(N_NODES*EMB + 255)/256, 256, 0, stream>>>(hprev, batch, POOL);
  readout_kernel<<<N_GRAPHS, EMB, 0, stream>>>(POOL, p1w, p1b, p2w, p2b, (float*)d_out);
}
